// GearNet_Edge_30889404793309
// MI455X (gfx1250) — matrix operand order repaired and checked by reading, not compiled, not run on hardware
//
#include <hip/hip_runtime.h>
#include <hip/hip_bf16.h>

// ---------------------------------------------------------------------------
// GearNet-Edge forward for MI455X (gfx1250, wave32, WMMA).
// Dense einsum contractions run on v_wmma_f32_16x16x32_bf16 (f32 accumulate).
// A-side data is pre-converted to bf16 so the GEMM inner loop is pure
// b128-load + wmma (no per-iteration VALU conversion competing with XDL).
// ---------------------------------------------------------------------------

#define N_NODES 15000
#define N_EDGES 50000
#define N_MSG   200000
#define N_GRAPH 64
#define R_NODE  7
#define R_EDGE  8
#define BN_EPS  1e-5f

typedef __attribute__((ext_vector_type(16))) __bf16 v16bf;
typedef __attribute__((ext_vector_type(8)))  float  v8f;

union BF16x16 {
    v16bf v;
    unsigned short u[16];
    uint4 q[2];
};

__device__ __forceinline__ unsigned short f2bf(float f) {
    // round-to-nearest-even f32 -> bf16
    unsigned int u = __float_as_uint(f);
    unsigned int r = u + 0x7FFFu + ((u >> 16) & 1u);
    return (unsigned short)(r >> 16);
}

// ---------------------------------------------------------------------------
// Bulk f32 -> bf16 conversion (4 elements / thread, float4 in, uint2 out).
__global__ void cvt_bf16(const float* __restrict__ src, unsigned short* __restrict__ dst,
                         size_t n4) {
    size_t t = (size_t)blockIdx.x * blockDim.x + threadIdx.x;
    if (t >= n4) return;
    float4 v = *(const float4*)(src + t * 4);
    unsigned int lo = (unsigned int)f2bf(v.x) | ((unsigned int)f2bf(v.y) << 16);
    unsigned int hi = (unsigned int)f2bf(v.z) | ((unsigned int)f2bf(v.w) << 16);
    uint2 o; o.x = lo; o.y = hi;
    *(uint2*)(dst + t * 4) = o;
}

// ---------------------------------------------------------------------------
// Copy [rows, inC] f32 into [rows, Kp] with zero padding.
__global__ void pad_copy(const float* __restrict__ src, float* __restrict__ dst,
                         int rows, int inC, int Kp) {
    size_t t = (size_t)blockIdx.x * blockDim.x + threadIdx.x;
    if (t >= (size_t)rows * Kp) return;
    int c = (int)(t % Kp);
    size_t r = t / Kp;
    dst[t] = (c < inC) ? src[r * inC + c] : 0.f;
}

// ---------------------------------------------------------------------------
// Pack f32 weights [R, K, Nout] into per-lane bf16 WMMA B-fragment layout:
// blocks indexed [r][ko][no], each block = 32 lanes * 16 bf16, lane l holds
// B[ko*32 + 8*(l/16) + (j%8) + 16*(j/8)][no*16 + (l%16)], K zero-padded to Kp.
__global__ void pack_w(const float* __restrict__ W, unsigned short* __restrict__ out,
                       int R, int K, int Kp, int Nout) {
    int KT = Kp >> 5, NT = Nout >> 4;
    size_t total = (size_t)R * KT * NT * 512;
    size_t t = (size_t)blockIdx.x * blockDim.x + threadIdx.x;
    if (t >= total) return;
    int j = (int)(t & 15);
    int l = (int)((t >> 4) & 31);
    size_t blk = t >> 9;
    int no = (int)(blk % NT);
    size_t rk = blk / NT;
    int ko = (int)(rk % KT);
    int r  = (int)(rk / KT);
    int k = ko * 32 + ((l >> 4) << 3) + (j & 7) + ((j >> 3) << 4);
    int n = no * 16 + (l & 15);
    float val = (k < K) ? W[((size_t)r * K + k) * Nout + n] : 0.f;
    out[t] = f2bf(val);
}

// ---------------------------------------------------------------------------
// Scatter-add: agg[(dst*R + rel), :] += feat[src, :]   (f32 atomics, float4 reads)
__global__ void scatter_agg(const float* __restrict__ feat,
                            const int* __restrict__ srcIdx, const int* __restrict__ dstIdx,
                            const int* __restrict__ rel,
                            float* __restrict__ agg, int nEdges, int Kp, int R) {
    int chunks = Kp >> 2;
    size_t t = (size_t)blockIdx.x * blockDim.x + threadIdx.x;
    if (t >= (size_t)nEdges * chunks) return;
    int e = (int)(t / chunks);
    int c = (int)(t % chunks) << 2;
    int s = srcIdx[e];
    size_t seg = (size_t)dstIdx[e] * R + rel[e];
    float4 v = *(const float4*)(feat + (size_t)s * Kp + c);
    float* p = agg + seg * Kp + c;
    atomicAdd(p + 0, v.x);
    atomicAdd(p + 1, v.y);
    atomicAdd(p + 2, v.z);
    atomicAdd(p + 3, v.w);
}

// Scatter-add edge messages into node features: h[dst, :] += em[e, :]
__global__ void scatter_msg(const float* __restrict__ em, const int* __restrict__ dstIdx,
                            float* __restrict__ h, int nEdges, int C) {
    int chunks = C >> 2;
    size_t t = (size_t)blockIdx.x * blockDim.x + threadIdx.x;
    if (t >= (size_t)nEdges * chunks) return;
    int e = (int)(t / chunks);
    int c = (int)(t % chunks) << 2;
    float4 v = *(const float4*)(em + (size_t)e * C + c);
    float* p = h + (size_t)dstIdx[e] * C + c;
    atomicAdd(p + 0, v.x);
    atomicAdd(p + 1, v.y);
    atomicAdd(p + 2, v.z);
    atomicAdd(p + 3, v.w);
}

// ---------------------------------------------------------------------------
// RGCN GEMM: out[M, Nout] = sum_r agg[:, r, :] @ W_r  +  xin @ Ws  +  bias
// agg: [M, R, Kp] bf16; xin: [M, Kp] bf16; Wp/Wsp: packed bf16 fragments.
// Block: 256 threads (8 wave32). Block tile 128(M) x 256(N).
// Wave (w%4) selects 32-row M slice, (w/4) selects 128-col N slice.
// Per wave: 2 A-frags x 8 B-frags -> 16 wmma per 32-K chunk, C in 128 VGPRs.
__global__ __launch_bounds__(256) void rgcn_gemm(
    const unsigned short* __restrict__ agg, const unsigned short* __restrict__ xin,
    const unsigned short* __restrict__ Wp, const unsigned short* __restrict__ Wsp,
    const float* __restrict__ bias, float* __restrict__ out,
    int M, int Kp, int Nout, int R, int act) {
    const int tid = threadIdx.x;
    const int wv = tid >> 5;
    const int lane = tid & 31;
    const int halfsel = lane >> 4;
    const int mlane = lane & 15;
    const int mBase = blockIdx.x * 128 + (wv & 3) * 32;
    const int nBase = blockIdx.y * 256 + (wv >> 2) * 128;
    const int KT = Kp >> 5;
    const int NT16 = Nout >> 4;

    int row0 = mBase + mlane;       if (row0 >= M) row0 = M - 1;
    int row1 = mBase + 16 + mlane;  if (row1 >= M) row1 = M - 1;

    v8f c[2][8];
    const v8f vz = {0.f, 0.f, 0.f, 0.f, 0.f, 0.f, 0.f, 0.f};
#pragma unroll
    for (int a = 0; a < 2; ++a)
#pragma unroll
        for (int b = 0; b < 8; ++b) c[a][b] = vz;

    for (int rr = 0; rr <= R; ++rr) {
        const unsigned short* a0;
        const unsigned short* a1;
        const unsigned short* wb;
        if (rr < R) {
            a0 = agg + ((size_t)row0 * R + rr) * Kp;
            a1 = agg + ((size_t)row1 * R + rr) * Kp;
            wb = Wp + (size_t)rr * KT * NT16 * 512;
        } else {
            a0 = xin + (size_t)row0 * Kp;
            a1 = xin + (size_t)row1 * Kp;
            wb = Wsp;
        }
        for (int ko = 0; ko < KT; ++ko) {
            // A-fragment: lane l holds K runs [k0..k0+7], [k0+16..k0+23],
            // k0 = ko*32 + 8*(l/16); bf16 in memory -> two b128 loads.
            const int kchunk = ko * 32 + (halfsel << 3);
            BF16x16 af0, af1;
            af0.q[0] = *(const uint4*)(a0 + kchunk);
            af0.q[1] = *(const uint4*)(a0 + kchunk + 16);
            af1.q[0] = *(const uint4*)(a1 + kchunk);
            af1.q[1] = *(const uint4*)(a1 + kchunk + 16);
#pragma unroll
            for (int nt = 0; nt < 8; ++nt) {
                size_t no = (size_t)(nBase >> 4) + nt;
                BF16x16 bfr;
                const uint4* bp = (const uint4*)(wb + ((size_t)ko * NT16 + no) * 512 + lane * 16);
                bfr.q[0] = bp[0];
                bfr.q[1] = bp[1];
                c[0][nt] = __builtin_amdgcn_wmma_f32_16x16x32_bf16(
                    false, af0.v, false, bfr.v, (short)0, c[0][nt], false, false);
                c[1][nt] = __builtin_amdgcn_wmma_f32_16x16x32_bf16(
                    false, af1.v, false, bfr.v, (short)0, c[1][nt], false, false);
            }
        }
    }

    // Epilogue: D element v of lane l -> row = frag_m + 8*(l/16) + v, col = nt*16 + (l%16)
#pragma unroll
    for (int nt = 0; nt < 8; ++nt) {
        int col = nBase + nt * 16 + mlane;
        float bv = bias[col];
#pragma unroll
        for (int a = 0; a < 2; ++a) {
            int mrow = mBase + a * 16 + (halfsel << 3);
#pragma unroll
            for (int v = 0; v < 8; ++v) {
                int m = mrow + v;
                if (m < M) {
                    float val = c[a][nt][v] + bv;
                    if (act) val = fmaxf(val, 0.f);
                    out[(size_t)m * Nout + col] = val;
                }
            }
        }
    }
}

// ---------------------------------------------------------------------------
// BatchNorm column statistics (sum, sumsq) with atomics into stats[2*C].
__global__ void bn_stats(const float* __restrict__ h, float* __restrict__ stats,
                         int Nrows, int C) {
    int col = blockIdx.x * blockDim.x + threadIdx.x;
    if (col >= C) return;
    int chunk = (Nrows + gridDim.y - 1) / gridDim.y;
    int r0 = blockIdx.y * chunk;
    int r1 = r0 + chunk; if (r1 > Nrows) r1 = Nrows;
    float s = 0.f, s2 = 0.f;
    for (int r = r0; r < r1; ++r) {
        float v = h[(size_t)r * C + col];
        s += v;
        s2 += v * v;
    }
    atomicAdd(&stats[col], s);
    atomicAdd(&stats[C + col], s2);
}

__global__ void bn_apply(float* __restrict__ h, const float* __restrict__ stats,
                         const float* __restrict__ g, const float* __restrict__ b,
                         int Nrows, int C, int lrelu) {
    size_t t = (size_t)blockIdx.x * blockDim.x + threadIdx.x;
    if (t >= (size_t)Nrows * C) return;
    int col = (int)(t % C);
    float invN = 1.f / (float)Nrows;
    float mu = stats[col] * invN;
    float var = stats[C + col] * invN - mu * mu;
    float val = (h[t] - mu) * rsqrtf(var + BN_EPS) * g[col] + b[col];
    if (lrelu) val = (val > 0.f) ? val : 0.01f * val;
    h[t] = val;
}

// ---------------------------------------------------------------------------
// Mean-pool + projection head.
__global__ void pool_count(const int* __restrict__ batch, float* __restrict__ cnt, int Nrows) {
    int n = blockIdx.x * blockDim.x + threadIdx.x;
    if (n >= Nrows) return;
    atomicAdd(&cnt[batch[n]], 1.f);
}

__global__ void pool_sum(const float* __restrict__ h, const int* __restrict__ batch,
                         float* __restrict__ pool, int Nrows, int C) {
    size_t t = (size_t)blockIdx.x * blockDim.x + threadIdx.x;
    if (t >= (size_t)Nrows * C) return;
    int n = (int)(t / C);
    int c = (int)(t % C);
    atomicAdd(&pool[(size_t)batch[n] * C + c], h[t]);
}

__global__ void head1(const float* __restrict__ pool, const float* __restrict__ cnt,
                      const float* __restrict__ W, const float* __restrict__ b,
                      float* __restrict__ out) {
    int t = blockIdx.x * blockDim.x + threadIdx.x;
    if (t >= N_GRAPH * 300) return;
    int g = t / 300, j = t % 300;
    float inv = 1.f / fmaxf(cnt[g], 1.f);
    float acc = 0.f;
    for (int k = 0; k < 512; ++k) acc += pool[(size_t)g * 512 + k] * W[k * 300 + j];
    out[t] = fmaxf(acc * inv + b[j], 0.f);
}

__global__ void head2(const float* __restrict__ mid, const float* __restrict__ W,
                      const float* __restrict__ b, float* __restrict__ out) {
    int t = blockIdx.x * blockDim.x + threadIdx.x;
    if (t >= N_GRAPH * 300) return;
    int g = t / 300, j = t % 300;
    float acc = b[j];
    for (int k = 0; k < 300; ++k) acc += mid[g * 300 + k] * W[k * 300 + j];
    out[t] = acc;
}

// ---------------------------------------------------------------------------
extern "C" void kernel_launch(void* const* d_in, const int* in_sizes, int n_in,
                              void* d_out, int out_size, void* d_ws, size_t ws_size,
                              hipStream_t stream) {
    (void)in_sizes; (void)n_in; (void)out_size; (void)ws_size;

    const float* x          = (const float*)d_in[0];
    const int*   eidx       = (const int*)d_in[1];
    const int*   etype      = (const int*)d_in[2];
    const float* eattr      = (const float*)d_in[3];
    const int*   emidx      = (const int*)d_in[4];
    const int*   emrel      = (const int*)d_in[5];
    const int*   batch      = (const int*)d_in[6];
    const int* src  = eidx;             const int* dst  = eidx + N_EDGES;
    const int* esrc = emidx;            const int* edst = emidx + N_MSG;

    const int cin[6]  = {22, 256, 512, 1024, 1024, 1024};
    const int cout_[6]= {256, 512, 1024, 1024, 1024, 512};
    const int ein[6]  = {53, 256, 512, 1024, 1024, 1024};
    int kpN[6], kpE[6];
    for (int i = 0; i < 6; ++i) {
        kpN[i] = (cin[i] + 31) & ~31;
        kpE[i] = (ein[i] + 31) & ~31;
    }

    // ---- workspace layout -------------------------------------------------
    char* ws = (char*)d_ws;
    size_t off = 0;
    auto carve = [&](size_t bytes) -> char* {
        char* p = ws + off;
        off = (off + bytes + 255) & ~(size_t)255;
        return p;
    };
    float* aggE  = (float*)carve((size_t)N_EDGES * R_EDGE * 1024 * 4);
    float* aggN  = (float*)carve((size_t)N_NODES * R_NODE * 1024 * 4);
    unsigned short* aggEb = (unsigned short*)carve((size_t)N_EDGES * R_EDGE * 1024 * 2);
    unsigned short* aggNb = (unsigned short*)carve((size_t)N_NODES * R_NODE * 1024 * 2);
    float* emA   = (float*)carve((size_t)N_EDGES * 1024 * 4);
    float* emB   = (float*)carve((size_t)N_EDGES * 1024 * 4);
    unsigned short* emBf = (unsigned short*)carve((size_t)N_EDGES * 1024 * 2);
    float* hA    = (float*)carve((size_t)N_NODES * 1024 * 4);
    float* hB    = (float*)carve((size_t)N_NODES * 1024 * 4);
    unsigned short* hBf  = (unsigned short*)carve((size_t)N_NODES * 1024 * 2);
    float* stats = (float*)carve(2 * 1024 * 4);
    float* pool  = (float*)carve((size_t)N_GRAPH * 512 * 4);
    float* cnt   = (float*)carve((size_t)N_GRAPH * 4);
    float* mid   = (float*)carve((size_t)N_GRAPH * 300 * 4);
    unsigned short *pcW[6], *pcS[6], *peW[6], *peS[6];
    for (int i = 0; i < 6; ++i) {
        pcW[i] = (unsigned short*)carve((size_t)R_NODE * kpN[i] * cout_[i] * 2);
        pcS[i] = (unsigned short*)carve((size_t)kpN[i] * cout_[i] * 2);
        peW[i] = (unsigned short*)carve((size_t)R_EDGE * kpE[i] * cout_[i] * 2);
        peS[i] = (unsigned short*)carve((size_t)kpE[i] * cout_[i] * 2);
    }

    auto cdiv = [](size_t a, size_t b) { return (a + b - 1) / b; };

    // ---- pack weights to bf16 B-fragment layout, pad inputs ---------------
    pad_copy<<<(int)cdiv((size_t)N_NODES * kpN[0], 256), 256, 0, stream>>>(x, hA, N_NODES, 22, kpN[0]);
    pad_copy<<<(int)cdiv((size_t)N_EDGES * kpE[0], 256), 256, 0, stream>>>(eattr, emA, N_EDGES, 53, kpE[0]);
    for (int i = 0; i < 6; ++i) {
        const float* cW = (const float*)d_in[7 + 3 * i];
        const float* cS = (const float*)d_in[8 + 3 * i];
        const float* eW = (const float*)d_in[25 + 3 * i];
        const float* eS = (const float*)d_in[26 + 3 * i];
        size_t t;
        t = (size_t)R_NODE * (kpN[i] >> 5) * (cout_[i] >> 4) * 512;
        pack_w<<<(int)cdiv(t, 256), 256, 0, stream>>>(cW, pcW[i], R_NODE, cin[i], kpN[i], cout_[i]);
        t = (size_t)(kpN[i] >> 5) * (cout_[i] >> 4) * 512;
        pack_w<<<(int)cdiv(t, 256), 256, 0, stream>>>(cS, pcS[i], 1, cin[i], kpN[i], cout_[i]);
        t = (size_t)R_EDGE * (kpE[i] >> 5) * (cout_[i] >> 4) * 512;
        pack_w<<<(int)cdiv(t, 256), 256, 0, stream>>>(eW, peW[i], R_EDGE, ein[i], kpE[i], cout_[i]);
        t = (size_t)(kpE[i] >> 5) * (cout_[i] >> 4) * 512;
        pack_w<<<(int)cdiv(t, 256), 256, 0, stream>>>(eS, peS[i], 1, ein[i], kpE[i], cout_[i]);
    }

    // ---- layer loop -------------------------------------------------------
    float* emIn = emA; float* emOut = emB;
    float* hIn  = hA;  float* hOut  = hB;
    for (int i = 0; i < 6; ++i) {
        const float* cb = (const float*)d_in[9 + 3 * i];
        const float* eb = (const float*)d_in[27 + 3 * i];
        const float* bg; const float* bb;
        if (cout_[i] == 256)      { bg = (const float*)d_in[43]; bb = (const float*)d_in[44]; }
        else if (cout_[i] == 512) { bg = (const float*)d_in[45]; bb = (const float*)d_in[46]; }
        else                      { bg = (const float*)d_in[47]; bb = (const float*)d_in[48]; }
        int C = cout_[i];

        // 1) edge-branch RGCN: em = relu(agg ⊗ eW + em @ eS + eb)
        hipMemsetAsync(aggE, 0, (size_t)N_EDGES * R_EDGE * kpE[i] * 4, stream);
        scatter_agg<<<(int)cdiv((size_t)N_MSG * (kpE[i] >> 2), 256), 256, 0, stream>>>(
            emIn, esrc, edst, emrel, aggE, N_MSG, kpE[i], R_EDGE);
        cvt_bf16<<<(int)cdiv((size_t)N_EDGES * R_EDGE * kpE[i] / 4, 256), 256, 0, stream>>>(
            aggE, aggEb, (size_t)N_EDGES * R_EDGE * kpE[i] / 4);
        cvt_bf16<<<(int)cdiv((size_t)N_EDGES * kpE[i] / 4, 256), 256, 0, stream>>>(
            emIn, emBf, (size_t)N_EDGES * kpE[i] / 4);
        {
            dim3 grid((unsigned)cdiv(N_EDGES, 128), (unsigned)(C / 256));
            rgcn_gemm<<<grid, 256, 0, stream>>>(aggEb, emBf, peW[i], peS[i], eb, emOut,
                                                N_EDGES, kpE[i], C, R_EDGE, 1);
        }

        // 2) node-branch RGCN: h = agg ⊗ cW + h @ cS + cb + scatter(emOut)
        hipMemsetAsync(aggN, 0, (size_t)N_NODES * R_NODE * kpN[i] * 4, stream);
        scatter_agg<<<(int)cdiv((size_t)N_EDGES * (kpN[i] >> 2), 256), 256, 0, stream>>>(
            hIn, src, dst, etype, aggN, N_EDGES, kpN[i], R_NODE);
        cvt_bf16<<<(int)cdiv((size_t)N_NODES * R_NODE * kpN[i] / 4, 256), 256, 0, stream>>>(
            aggN, aggNb, (size_t)N_NODES * R_NODE * kpN[i] / 4);
        cvt_bf16<<<(int)cdiv((size_t)N_NODES * kpN[i] / 4, 256), 256, 0, stream>>>(
            hIn, hBf, (size_t)N_NODES * kpN[i] / 4);
        {
            dim3 grid((unsigned)cdiv(N_NODES, 128), (unsigned)(C / 256));
            rgcn_gemm<<<grid, 256, 0, stream>>>(aggNb, hBf, pcW[i], pcS[i], cb, hOut,
                                                N_NODES, kpN[i], C, R_NODE, 0);
        }
        scatter_msg<<<(int)cdiv((size_t)N_EDGES * (C >> 2), 256), 256, 0, stream>>>(
            emOut, dst, hOut, N_EDGES, C);

        // 3) BatchNorm (+ leaky relu for layers 0..4)
        hipMemsetAsync(stats, 0, 2 * (size_t)C * 4, stream);
        {
            dim3 grid((unsigned)cdiv(C, 256), 60);
            bn_stats<<<grid, 256, 0, stream>>>(hOut, stats, N_NODES, C);
        }
        bn_apply<<<(int)cdiv((size_t)N_NODES * C, 256), 256, 0, stream>>>(
            hOut, stats, bg, bb, N_NODES, C, (i < 5) ? 1 : 0);

        // swap ping-pong buffers
        float* t;
        t = emIn; emIn = emOut; emOut = t;
        t = hIn;  hIn  = hOut;  hOut  = t;
    }

    // ---- global mean pool + projection head -------------------------------
    hipMemsetAsync(pool, 0, (size_t)N_GRAPH * 512 * 4, stream);
    hipMemsetAsync(cnt, 0, (size_t)N_GRAPH * 4, stream);
    pool_count<<<(int)cdiv(N_NODES, 256), 256, 0, stream>>>(batch, cnt, N_NODES);
    pool_sum<<<(int)cdiv((size_t)N_NODES * 512, 256), 256, 0, stream>>>(hIn, batch, pool, N_NODES, 512);

    const float* hW1 = (const float*)d_in[49];
    const float* hb1 = (const float*)d_in[50];
    const float* hW2 = (const float*)d_in[51];
    const float* hb2 = (const float*)d_in[52];
    head1<<<(int)cdiv((size_t)N_GRAPH * 300, 256), 256, 0, stream>>>(pool, cnt, hW1, hb1, mid);
    head2<<<(int)cdiv((size_t)N_GRAPH * 300, 256), 256, 0, stream>>>(mid, hW2, hb2, (float*)d_out);
}
